// Gamarjoba_24137716203926
// MI455X (gfx1250) — compile-verified, matrix operand order checked by reading, not run on hardware
//
#include <hip/hip_runtime.h>

// ---------------------------------------------------------------------------
// Types for CDNA5 WMMA (wave32, v_wmma_f32_16x16x32_bf16)
// ---------------------------------------------------------------------------
typedef __bf16 bf16;
typedef bf16  v16bf __attribute__((ext_vector_type(16)));
typedef float v8f   __attribute__((ext_vector_type(8)));

union FragB16 {            // 16 bf16 lane-fragment = two 16B chunks in memory
    v16bf v;
    struct { uint4 lo, hi; } q;
};
union B16x8 {              // 8 bf16 loaded as one b128
    uint4 q;
    bf16  e[8];
};

// ---------------------------------------------------------------------------
// Utility kernels
// ---------------------------------------------------------------------------
__global__ void cast_f32_bf16(const float* __restrict__ in,
                              bf16* __restrict__ out, size_t n) {
    size_t i = (size_t)blockIdx.x * blockDim.x + threadIdx.x;
    size_t stride = (size_t)gridDim.x * blockDim.x;
    for (; i < n; i += stride) out[i] = (bf16)in[i];
}

__global__ void add_vec(const float* __restrict__ a, const float* __restrict__ b,
                        float* __restrict__ o, int n) {
    int i = blockIdx.x * blockDim.x + threadIdx.x;
    int stride = gridDim.x * blockDim.x;
    for (; i < n; i += stride) o[i] = a[i] + b[i];
}

__global__ void zero_f32(float* __restrict__ p, int n) {
    int i = blockIdx.x * blockDim.x + threadIdx.x;
    int stride = gridDim.x * blockDim.x;
    for (; i < n; i += stride) p[i] = 0.0f;
}

// enc_x[s, :] = bf16(enc_emb[token[s], :])
__global__ void embed_enc(const int* __restrict__ tok,
                          const float* __restrict__ emb,
                          bf16* __restrict__ x, int H) {
    const int s = blockIdx.x;
    const int t = tok[s];
    const float* row = emb + (size_t)t * H;
    bf16* out = x + (size_t)s * H;
    for (int j = threadIdx.x; j < H; j += blockDim.x) out[j] = (bf16)row[j];
}

// dec_x[s, :] = bf16(relu(dec_emb[s==0 ? 0 : target[s-1], :]))
__global__ void embed_dec(const int* __restrict__ target,
                          const float* __restrict__ emb,
                          bf16* __restrict__ x, int H) {
    const int s = blockIdx.x;
    const int t = (s == 0) ? 0 : target[s - 1];
    const float* row = emb + (size_t)t * H;
    bf16* out = x + (size_t)s * H;
    for (int j = threadIdx.x; j < H; j += blockDim.x) {
        float v = row[j];
        out[j] = (bf16)(v > 0.0f ? v : 0.0f);
    }
}

// ---------------------------------------------------------------------------
// WMMA GEMM:  C[n, m] = sum_k A[m, k] * B[n, k]  (+ bias[m])
//   A: [M, K] bf16 row-major, B: [N, K] bf16 row-major, C: [N, ldC] fp32.
// One wave (32 threads) per block computes a 16(M) x 128(N) strip:
//   - one A fragment amortized over 8 WMMAs (A re-read factor = N/128)
//   - all 8 B fragments loaded before the WMMA burst so the compiler can
//     clause the loads and overlap later loads with earlier WMMAs.
// Fragment layout follows the CDNA5 ISA 16-bit A/B tables: each lane holds
// two contiguous 16-byte K-chunks, so loads are plain b128s.
// ---------------------------------------------------------------------------
#define NSUB 8   // 8 x 16 = 128-wide N strip per wave

__global__ void __launch_bounds__(32)
gemm_bf16_nt_wmma(const bf16* __restrict__ A, const bf16* __restrict__ B,
                  const float* __restrict__ bias, float* __restrict__ C,
                  int M, int N, int K, int ldC) {
    const int lane  = threadIdx.x;       // exactly one wave32; EXEC all-ones
    const int li    = lane & 15;         // A row within tile / B column within tile
    const int g     = lane >> 4;         // K sub-chunk selector
    const int mtile = blockIdx.x;
    const int nbase = blockIdx.y * (16 * NSUB);

    int m_frag = mtile * 16 + li;        // clamp loads on ragged M edge (stores guarded)
    if (m_frag >= M) m_frag = M - 1;
    const bf16* arow = A + (size_t)m_frag * K;

    v8f acc[NSUB];
    const v8f vz = {0.f, 0.f, 0.f, 0.f, 0.f, 0.f, 0.f, 0.f};
#pragma unroll
    for (int j = 0; j < NSUB; ++j) acc[j] = vz;

    for (int k0 = 0; k0 < K; k0 += 32) {
        FragB16 fa;
        fa.q.lo = *(const uint4*)(arow + k0 + g * 8);
        fa.q.hi = *(const uint4*)(arow + k0 + 16 + g * 8);
        if (k0 + 32 < K) __builtin_prefetch(arow + k0 + 32, 0, 1);

        FragB16 fb[NSUB];
#pragma unroll
        for (int j = 0; j < NSUB; ++j) {
            const bf16* brow = B + (size_t)(nbase + j * 16 + li) * K + k0;
            fb[j].q.lo = *(const uint4*)(brow + g * 8);
            fb[j].q.hi = *(const uint4*)(brow + 16 + g * 8);
        }
#pragma unroll
        for (int j = 0; j < NSUB; ++j) {
            acc[j] = __builtin_amdgcn_wmma_f32_16x16x32_bf16(
                         false, fa.v, false, fb[j].v, (short)0, acc[j], false, false);
        }
    }

    // C/D layout: lanes 0-15 -> N=li, rows r; lanes 16-31 -> rows r+8.
    const int mrow_off = g * 8;
#pragma unroll
    for (int j = 0; j < NSUB; ++j) {
        const int n_g = nbase + j * 16 + li;
        if (n_g >= N) continue;
#pragma unroll
        for (int r = 0; r < 8; ++r) {
            const int m_g = mtile * 16 + mrow_off + r;
            if (m_g < M) {
                float v = acc[j][r];
                if (bias) v += bias[m_g];
                C[(size_t)n_g * ldC + m_g] = v;
            }
        }
    }
}

// ---------------------------------------------------------------------------
// One LSTM layer-step (sequential part). 8-way K-split per hidden unit.
//   gates[m] = base[m] + Whh[m,:]·h  (+ Wih[m,:]·x  when Wih != null)
// base = precomputed input gates (layer 0) or combined bias (layer 1).
// grid = H/32 blocks of 256 threads = 8192 threads; weights stream from L2.
// ---------------------------------------------------------------------------
#define HDIM   1024
#define UNITS  32
#define KSPLIT 8
#define KSEG   (HDIM / KSPLIT)   // 128

__global__ void __launch_bounds__(256)
lstm_step(const float* __restrict__ base,
          const bf16*  __restrict__ Whh,
          const bf16*  __restrict__ Wih,    // null for layer 0
          const float* __restrict__ h_in,
          const float* __restrict__ c_in,
          const float* __restrict__ x_in,   // null for layer 0
          float* __restrict__ h_out,
          float* __restrict__ c_out,
          bf16*  __restrict__ top_out) {    // null unless decoder top layer
    __shared__ float sh_h[HDIM];
    __shared__ float sh_x[HDIM];
    __shared__ float red[UNITS][4][KSPLIT];

    const int tid = threadIdx.x;
    for (int i = tid; i < HDIM; i += 256) {
        sh_h[i] = h_in[i];
        sh_x[i] = x_in ? x_in[i] : 0.0f;
    }
    __syncthreads();

    const int u    = tid >> 3;               // hidden unit within block
    const int p    = tid & 7;                // K-split part
    const int j    = blockIdx.x * UNITS + u; // global hidden unit
    const int kbeg = p * KSEG;

#pragma unroll
    for (int gate = 0; gate < 4; ++gate) {
        const size_t row = (size_t)gate * HDIM + j;
        const bf16* wh = Whh + row * HDIM + kbeg;
        float s = 0.0f;
        for (int k8 = 0; k8 < KSEG; k8 += 8) {
            B16x8 w; w.q = *(const uint4*)(wh + k8);
#pragma unroll
            for (int e = 0; e < 8; ++e) s += (float)w.e[e] * sh_h[kbeg + k8 + e];
        }
        if (Wih) {
            const bf16* wi = Wih + row * HDIM + kbeg;
            for (int k8 = 0; k8 < KSEG; k8 += 8) {
                B16x8 w; w.q = *(const uint4*)(wi + k8);
#pragma unroll
                for (int e = 0; e < 8; ++e) s += (float)w.e[e] * sh_x[kbeg + k8 + e];
            }
        }
        red[u][gate][p] = s;
    }
    __syncthreads();

    if (p == 0) {
        float gs[4];
#pragma unroll
        for (int gate = 0; gate < 4; ++gate) {
            float s = base[gate * HDIM + j];
#pragma unroll
            for (int q = 0; q < KSPLIT; ++q) s += red[u][gate][q];
            gs[gate] = s;
        }
        const float ig = 1.0f / (1.0f + expf(-gs[0]));
        const float fg = 1.0f / (1.0f + expf(-gs[1]));
        const float gg = tanhf(gs[2]);
        const float og = 1.0f / (1.0f + expf(-gs[3]));
        const float cn = fg * c_in[j] + ig * gg;
        const float hn = og * tanhf(cn);
        c_out[j] = cn;
        h_out[j] = hn;
        if (top_out) top_out[j] = (bf16)hn;
    }
}

// ---------------------------------------------------------------------------
// In-place row-wise log_softmax over V columns (one block per row).
// ---------------------------------------------------------------------------
__global__ void __launch_bounds__(256)
log_softmax_rows(float* __restrict__ logits, int V) {
    __shared__ float red[256];
    const int tid = threadIdx.x;
    float* row = logits + (size_t)blockIdx.x * V;

    float m = -3.4e38f;
    for (int i = tid; i < V; i += 256) m = fmaxf(m, row[i]);
    red[tid] = m; __syncthreads();
    for (int s = 128; s > 0; s >>= 1) {
        if (tid < s) red[tid] = fmaxf(red[tid], red[tid + s]);
        __syncthreads();
    }
    m = red[0]; __syncthreads();

    float sum = 0.0f;
    for (int i = tid; i < V; i += 256) sum += expf(row[i] - m);
    red[tid] = sum; __syncthreads();
    for (int s = 128; s > 0; s >>= 1) {
        if (tid < s) red[tid] += red[tid + s];
        __syncthreads();
    }
    const float lse = logf(red[0]) + m;

    for (int i = tid; i < V; i += 256) row[i] -= lse;
}

// ---------------------------------------------------------------------------
// Host orchestration
// ---------------------------------------------------------------------------
extern "C" void kernel_launch(void* const* d_in, const int* in_sizes, int n_in,
                              void* d_out, int out_size, void* d_ws, size_t ws_size,
                              hipStream_t stream) {
    (void)in_sizes; (void)n_in; (void)out_size; (void)ws_size;

    const int*   input_tensor = (const int*)d_in[0];
    const int*   target       = (const int*)d_in[1];
    const float* enc_emb = (const float*)d_in[2];
    const float* enc_Wih = (const float*)d_in[3];
    const float* enc_Whh = (const float*)d_in[4];
    const float* enc_bih = (const float*)d_in[5];
    const float* enc_bhh = (const float*)d_in[6];
    const float* dec_emb = (const float*)d_in[7];
    const float* dec_Wih = (const float*)d_in[8];
    const float* dec_Whh = (const float*)d_in[9];
    const float* dec_bih = (const float*)d_in[10];
    const float* dec_bhh = (const float*)d_in[11];
    const float* out_W   = (const float*)d_in[12];
    const float* out_b   = (const float*)d_in[13];
    float* out = (float*)d_out;

    const int V = 50257, H = 1024, L = 2, S = 512, T = 512;
    const int G = 4 * H;                  // 4096 gate rows
    const size_t WSZ = (size_t)G * H;     // elements per weight matrix

    // ---- carve workspace (~190 MB) ----
    uint8_t* ws = (uint8_t*)d_ws;
    size_t off = 0;
    auto carve = [&](size_t bytes) -> void* {
        void* p = ws + off;
        off += bytes;
        off = (off + 255) & ~(size_t)255;
        return p;
    };
    bf16*  encWih_b = (bf16*)carve((size_t)L * WSZ * 2);
    bf16*  encWhh_b = (bf16*)carve((size_t)L * WSZ * 2);
    bf16*  decWih_b = (bf16*)carve((size_t)L * WSZ * 2);
    bf16*  decWhh_b = (bf16*)carve((size_t)L * WSZ * 2);
    bf16*  outW_b   = (bf16*)carve((size_t)V * H * 2);
    bf16*  enc_x    = (bf16*)carve((size_t)S * H * 2);
    bf16*  dec_x    = (bf16*)carve((size_t)T * H * 2);
    float* gates_enc = (float*)carve((size_t)S * G * 4);
    float* gates_dec = (float*)carve((size_t)T * G * 4);
    bf16*  tops      = (bf16*)carve((size_t)T * H * 2);
    float* bias_c    = (float*)carve((size_t)4 * G * 4);   // encl0,encl1,decl0,decl1
    float* hbuf      = (float*)carve((size_t)2 * L * H * 4);
    float* cbuf      = (float*)carve((size_t)2 * L * H * 4);

    // ---- weight precision conversion (bf16 storage, fp32 accumulate) ----
    cast_f32_bf16<<<2048, 256, 0, stream>>>(enc_Wih, encWih_b, (size_t)L * WSZ);
    cast_f32_bf16<<<2048, 256, 0, stream>>>(enc_Whh, encWhh_b, (size_t)L * WSZ);
    cast_f32_bf16<<<2048, 256, 0, stream>>>(dec_Wih, decWih_b, (size_t)L * WSZ);
    cast_f32_bf16<<<2048, 256, 0, stream>>>(dec_Whh, decWhh_b, (size_t)L * WSZ);
    cast_f32_bf16<<<4096, 256, 0, stream>>>(out_W,   outW_b,   (size_t)V * H);

    // ---- combined biases (bih + bhh) per layer ----
    add_vec<<<16, 256, 0, stream>>>(enc_bih,     enc_bhh,     bias_c,         G);
    add_vec<<<16, 256, 0, stream>>>(enc_bih + G, enc_bhh + G, bias_c + G,     G);
    add_vec<<<16, 256, 0, stream>>>(dec_bih,     dec_bhh,     bias_c + 2 * G, G);
    add_vec<<<16, 256, 0, stream>>>(dec_bih + G, dec_bhh + G, bias_c + 3 * G, G);

    // ---- embeddings ----
    embed_enc<<<S, 256, 0, stream>>>(input_tensor, enc_emb, enc_x, H);
    embed_dec<<<T, 256, 0, stream>>>(target,       dec_emb, dec_x, H);

    // ---- hoisted layer-0 input GEMMs: gates_pre[t, m] = Wih0[m,:]·x_t + b ----
    dim3 gg(G / 16, S / (16 * NSUB));
    gemm_bf16_nt_wmma<<<gg, 32, 0, stream>>>(encWih_b, enc_x, bias_c,
                                             gates_enc, G, S, H, G);
    gemm_bf16_nt_wmma<<<gg, 32, 0, stream>>>(decWih_b, dec_x, bias_c + 2 * G,
                                             gates_dec, G, T, H, G);

    // ---- zero initial h/c (ping-pong buffers) ----
    zero_f32<<<16, 256, 0, stream>>>(hbuf, 2 * L * H);
    zero_f32<<<16, 256, 0, stream>>>(cbuf, 2 * L * H);

    auto hp = [&](int par, int l) { return hbuf + ((size_t)par * L + l) * H; };
    auto cp = [&](int par, int l) { return cbuf + ((size_t)par * L + l) * H; };
    const dim3 lgrid(H / UNITS);

    // ---- encoder recurrence ----
    for (int t = 0; t < S; ++t) {
        const int pb = t & 1, nb = pb ^ 1;
        lstm_step<<<lgrid, 256, 0, stream>>>(
            gates_enc + (size_t)t * G, encWhh_b, nullptr,
            hp(pb, 0), cp(pb, 0), nullptr, hp(nb, 0), cp(nb, 0), nullptr);
        lstm_step<<<lgrid, 256, 0, stream>>>(
            bias_c + G, encWhh_b + WSZ, encWih_b + WSZ,
            hp(pb, 1), cp(pb, 1), hp(nb, 0), hp(nb, 1), cp(nb, 1), nullptr);
    }

    // ---- decoder recurrence (state carries over; parity continues) ----
    for (int t = 0; t < T; ++t) {
        const int pb = t & 1, nb = pb ^ 1;
        lstm_step<<<lgrid, 256, 0, stream>>>(
            gates_dec + (size_t)t * G, decWhh_b, nullptr,
            hp(pb, 0), cp(pb, 0), nullptr, hp(nb, 0), cp(nb, 0), nullptr);
        lstm_step<<<lgrid, 256, 0, stream>>>(
            bias_c + 3 * G, decWhh_b + WSZ, decWih_b + WSZ,
            hp(pb, 1), cp(pb, 1), hp(nb, 0), hp(nb, 1), cp(nb, 1),
            tops + (size_t)t * H);
    }

    // ---- output projection: logits[t, v] = out_W[v,:]·top_t + out_b[v] ----
    dim3 gl((V + 15) / 16, T / (16 * NSUB));
    gemm_bf16_nt_wmma<<<gl, 32, 0, stream>>>(outW_b, tops, out_b, out, V, T, H, V);

    // ---- in-place log_softmax ----
    log_softmax_rows<<<T, 256, 0, stream>>>(out, V);
}
